// QuantizedLinear_4423816315410
// MI455X (gfx1250) — matrix operand order repaired and checked by reading, not compiled, not run on hardware
//
#include <hip/hip_runtime.h>
#include <stdint.h>

typedef __attribute__((ext_vector_type(16))) _Float16 v16h;
typedef __attribute__((ext_vector_type(8)))  _Float16 v8h;
typedef __attribute__((ext_vector_type(4)))  _Float16 v4h;
typedef __attribute__((ext_vector_type(8)))  float    v8f;
typedef __attribute__((ext_vector_type(4)))  float    v4f;
typedef __attribute__((ext_vector_type(4)))  int      v4i;

#define TOKENS 8192
#define IN_F   4096
#define OUT_F  4096
#define BM 128
#define BN 128
#define BK 32
#define LDS_STRIDE (BK + 8)   /* halves: 80 B/row, 16-B aligned, bank-conflict pad */

// ---- CDNA5 async global->LDS DMA (ASYNCcnt path), with sync fallback ------
#if defined(__has_builtin)
#  if __has_builtin(__builtin_amdgcn_global_load_async_to_lds_b128)
#    define USE_ASYNC 1
#  endif
#  if __has_builtin(__builtin_amdgcn_s_wait_asynccnt)
#    define HAVE_WAIT_ASYNC_BUILTIN 1
#  endif
#endif

typedef __attribute__((address_space(1))) v4i g_v4i;   // global int4
typedef __attribute__((address_space(3))) v4i l_v4i;   // LDS int4

__device__ __forceinline__ void cp16h(const _Float16* g, _Float16* l)
{
#ifdef USE_ASYNC
    // generic LDS pointer low 32 bits == LDS byte offset (aperture layout)
    g_v4i* g0 = (g_v4i*)(uintptr_t)g;
    g_v4i* g1 = (g_v4i*)(uintptr_t)(g + 8);
    l_v4i* l0 = (l_v4i*)(uint32_t)(uintptr_t)l;
    l_v4i* l1 = (l_v4i*)(uint32_t)(uintptr_t)(l + 8);
    __builtin_amdgcn_global_load_async_to_lds_b128(g0, l0, 0, 0);
    __builtin_amdgcn_global_load_async_to_lds_b128(g1, l1, 0, 0);
#else
    *(v8h*)l       = *(const v8h*)g;
    *(v8h*)(l + 8) = *(const v8h*)(g + 8);
#endif
}

__device__ __forceinline__ void async_wait0()
{
#ifdef USE_ASYNC
#  ifdef HAVE_WAIT_ASYNC_BUILTIN
    __builtin_amdgcn_s_wait_asynccnt(0);
#  else
    asm volatile("s_wait_asynccnt 0x0" ::: "memory");
#  endif
#endif
}

// ---------------------------------------------------------------------------
// Stage 1: sum(|W|) -> d_ws[0]
// ---------------------------------------------------------------------------
__global__ void absmean_reduce(const float* __restrict__ w, float* __restrict__ sum, int n)
{
    __shared__ float red[256];
    float s = 0.0f;
    for (int i = blockIdx.x * blockDim.x + threadIdx.x; i < n; i += gridDim.x * blockDim.x)
        s += fabsf(w[i]);
    red[threadIdx.x] = s;
    __syncthreads();
#pragma unroll
    for (int off = 128; off > 0; off >>= 1) {
        if (threadIdx.x < off) red[threadIdx.x] += red[threadIdx.x + off];
        __syncthreads();
    }
    if (threadIdx.x == 0) atomicAdd(sum, red[0]);
}

// ---------------------------------------------------------------------------
// Stage 2: Q = clip(round(W/gamma), -1, 1) stored as f16 (exact ternary)
// ---------------------------------------------------------------------------
__global__ void quantize_w(const float* __restrict__ w, const float* __restrict__ sum,
                           _Float16* __restrict__ q)
{
    const float gamma = (*sum) * (1.0f / 16777216.0f) + 1e-5f;  // N = 4096*4096
    const float r = 1.0f / gamma;
    const size_t i = ((size_t)blockIdx.x * blockDim.x + threadIdx.x) * 4;
    v4f v = *(const v4f*)(w + i);
    v4h qq;
#pragma unroll
    for (int e = 0; e < 4; ++e) {
        float t = rintf(v[e] * r);
        t = fminf(1.0f, fmaxf(-1.0f, t));
        qq[e] = (_Float16)t;
    }
    *(v4h*)(q + i) = qq;
}

// ---------------------------------------------------------------------------
// Stage 3: one-shot Dekker split  x = x_hi + x_lo  (both exact-in-f16 halves)
// ---------------------------------------------------------------------------
__global__ void split_x(const float* __restrict__ x,
                        _Float16* __restrict__ xhi, _Float16* __restrict__ xlo)
{
    const size_t i = ((size_t)blockIdx.x * blockDim.x + threadIdx.x) * 4;
    v4f v = *(const v4f*)(x + i);
    v4h hi, lo;
#pragma unroll
    for (int e = 0; e < 4; ++e) {
        _Float16 h = (_Float16)v[e];
        hi[e] = h;
        lo[e] = (_Float16)(v[e] - (float)h);   // exact residual of the rounding
    }
    *(v4h*)(xhi + i) = hi;
    *(v4h*)(xlo + i) = lo;
}

// ---------------------------------------------------------------------------
// Stage 4: out = (x_hi + x_lo) @ Q^T + bias, double-buffered async WMMA GEMM
//   unroll 1 on the K-loop: one loop body -> loop-carried acc phis coalesce
//   with in-place WMMA accumulation, killing the v_mov_b64 eviction chains.
// ---------------------------------------------------------------------------
__global__ __launch_bounds__(256, 1)
void ternary_gemm(const _Float16* __restrict__ xhi,
                  const _Float16* __restrict__ xlo,
                  const _Float16* __restrict__ q,
                  const float* __restrict__ bias,
                  float* __restrict__ out)
{
    __shared__ _Float16 sAhi[2][BM * LDS_STRIDE];
    __shared__ _Float16 sAlo[2][BM * LDS_STRIDE];
    __shared__ _Float16 sB  [2][BN * LDS_STRIDE];

    const int t    = threadIdx.x;
    const int lane = t & 31;
    const int wave = t >> 5;
    const int wm   = (wave >> 1) * 32;   // wave M offset (4 waves along M)
    const int wn   = (wave & 1) * 64;    // wave N offset (2 waves along N)
    const int gm   = blockIdx.y * BM;
    const int gn   = blockIdx.x * BN;

    const int lh = lane & 15;            // row/col within 16x16 tile
    const int lg = lane >> 4;            // lane group -> K chunk / M offset

    // loader indices: 2 threads per row, 16 halves (2 x b128) each
    const int ar   = t >> 1;
    const int ac   = (t & 1) * 16;
    const int ldso = ar * LDS_STRIDE + ac;

    const _Float16* hp = xhi + (size_t)(gm + ar) * IN_F + ac;
    const _Float16* lp = xlo + (size_t)(gm + ar) * IN_F + ac;
    const _Float16* qp = q   + (size_t)(gn + ar) * IN_F + ac;

    v8f acc[2][4] = {};

    // prologue: DMA tile 0 into stage 0
    cp16h(hp, &sAhi[0][ldso]);
    cp16h(lp, &sAlo[0][ldso]);
    cp16h(qp, &sB  [0][ldso]);

    union HU { v16h v; v8h h[2]; };

    const int KSTEPS = IN_F / BK;
#pragma unroll 1
    for (int ks = 0; ks < KSTEPS; ++ks) {
        const int s = ks & 1;
        async_wait0();                   // my stage-s DMAs have landed
        __syncthreads();                 // everyone's have landed

        if (ks + 1 < KSTEPS) {           // overlap next tile's DMA with WMMAs
            const int k1 = (ks + 1) * BK;
            cp16h(hp + k1, &sAhi[s ^ 1][ldso]);
            cp16h(lp + k1, &sAlo[s ^ 1][ldso]);
            cp16h(qp + k1, &sB  [s ^ 1][ldso]);
        }

        // ---- B fragments (shared by both M-subtiles): 32 VGPRs live ----
        HU bfr[4];
        const int kb = lg * 8;           // A: lane group -> K {0-7,16-23}/{8-15,24-31}
#pragma unroll
        for (int tn = 0; tn < 4; ++tn) { // B: lane = column, lane group -> K {0-7,16-23}/{8-15,24-31}
            const int col = (wn + tn * 16 + lh) * LDS_STRIDE + lg * 8;
            bfr[tn].h[0] = *(const v8h*)&sB[s][col];
            bfr[tn].h[1] = *(const v8h*)&sB[s][col + 16];
        }

        // ---- A fragments loaded per-tm: only 16 VGPRs of A live at once ----
#pragma unroll
        for (int tm = 0; tm < 2; ++tm) {
            const int row = (wm + tm * 16 + lh) * LDS_STRIDE;
            HU ahi, alo;
            ahi.h[0] = *(const v8h*)&sAhi[s][row + kb];
            ahi.h[1] = *(const v8h*)&sAhi[s][row + kb + 16];
            alo.h[0] = *(const v8h*)&sAlo[s][row + kb];
            alo.h[1] = *(const v8h*)&sAlo[s][row + kb + 16];
#pragma unroll
            for (int tn = 0; tn < 4; ++tn) {
                acc[tm][tn] = __builtin_amdgcn_wmma_f32_16x16x32_f16(
                    false, ahi.v, false, bfr[tn].v, (short)0, acc[tm][tn], false, false);
                acc[tm][tn] = __builtin_amdgcn_wmma_f32_16x16x32_f16(
                    false, alo.v, false, bfr[tn].v, (short)0, acc[tm][tn], false, false);
            }
        }
        // no trailing barrier: stage s^1 writers already synced at loop top;
        // stage s is only overwritten after the *next* top barrier.
    }

    // ---- epilogue: C/D layout VGPR r -> M = r + 8*lg, N = lh ----
#pragma unroll
    for (int tm = 0; tm < 2; ++tm) {
#pragma unroll
        for (int tn = 0; tn < 4; ++tn) {
            const int col = gn + wn + tn * 16 + lh;
            const float b = bias[col];
#pragma unroll
            for (int r = 0; r < 8; ++r) {
                const int row = gm + wm + tm * 16 + r + 8 * lg;
                out[(size_t)row * OUT_F + col] = acc[tm][tn][r] + b;
            }
        }
    }
}

// ---------------------------------------------------------------------------
extern "C" void kernel_launch(void* const* d_in, const int* in_sizes, int n_in,
                              void* d_out, int out_size, void* d_ws, size_t ws_size,
                              hipStream_t stream) {
    const float* x    = (const float*)d_in[0];   // [8192, 4096]
    const float* w    = (const float*)d_in[1];   // [4096, 4096]
    const float* bias = (const float*)d_in[2];   // [4096]
    float*       out  = (float*)d_out;

    char* ws = (char*)d_ws;
    float*    sum = (float*)ws;                                  // scalar |W| sum
    _Float16* q   = (_Float16*)(ws + 256);                       // 32 MB ternary Q
    _Float16* xhi = (_Float16*)(ws + 256 + (size_t)33554432);    // 64 MB
    _Float16* xlo = (_Float16*)(ws + 256 + (size_t)33554432 + 67108864); // 64 MB

    (void)hipMemsetAsync(d_ws, 0, 256, stream);                  // deterministic

    absmean_reduce<<<1024, 256, 0, stream>>>(w, sum, IN_F * OUT_F);

    quantize_w<<<(IN_F * OUT_F / 4) / 256, 256, 0, stream>>>(w, sum, q);

    split_x<<<((size_t)TOKENS * IN_F / 4) / 256, 256, 0, stream>>>(x, xhi, xlo);

    dim3 grid(OUT_F / BN, TOKENS / BM);                          // 32 x 64 blocks
    ternary_gemm<<<grid, 256, 0, stream>>>(xhi, xlo, q, bias, out);
}
